// StoryMemory_64201171140750
// MI455X (gfx1250) — compile-verified, matrix operand order checked by reading, not run on hardware
//
#include <hip/hip_runtime.h>

// ---------------- problem constants ----------------
constexpr int B_  = 16;
constexpr int T_  = 4096;
constexpr int D_  = 1024;
constexpr int S_  = 8;      // N_SLOTS
constexpr int C_  = 32;     // SLOT_DIM
constexpr int F_  = S_ * C_;        // 256
constexpr long long BT = (long long)B_ * T_;   // 65536 tokens
constexpr int NCH = 32;     // chunks per batch for the scan
constexpr int CH  = T_ / NCH;       // 128 steps per chunk
constexpr int P_  = 48;     // combined projection width: 8 gate + 32 val + 8 query

// WMMA fragment types (CDNA5, wave32)
typedef __attribute__((ext_vector_type(16))) __bf16 v16bf;
typedef __attribute__((ext_vector_type(8)))  float  v8f;

union V16 { v16bf v; unsigned int u[8]; unsigned short s[16]; };
union V8F { v8f v; float f[8]; };

// float -> bf16 via native cast (backend picks hw cvt, RNE)
__device__ __forceinline__ unsigned short f2bf(float f) {
  union { __bf16 h; unsigned short u; } cv;
  cv.h = (__bf16)f;
  return cv.u;
}
__device__ __forceinline__ unsigned int pack2(unsigned short lo, unsigned short hi) {
  return (unsigned int)lo | ((unsigned int)hi << 16);
}

// 16-bit A/B fragment K layout (ISA 7.12.2):
// VGPR i, lane-group kg (0: lanes 0-15, 1: lanes 16-31) holds K = kbase, kbase+1
__device__ __forceinline__ int kbase(int i, int kg) {
  return ((i & 4) ? 16 : 0) + (i & 3) * 2 + kg * 8;
}

// -------------------------------------------------------------------------
// Pack weights into WMMA B-fragment layout (bf16).
//   wcP : combined [W_write_gate; W_write_val; W_read_query] -> B[k][n]=W[n][k]
//         K=1024 (32 kb), N=48 (3 nb).  32*3*32*8 = 24576 words.
//   wroP: W_read_out as B[k=f][n=d] = Wro[d*256+f]. K=256 (8 kb), N=1024 (64 nb).
//         8*64*32*8 = 131072 words.
// word index = ((kb*NB + nb)*32 + lane)*8 + i ; word = {bf16(k), bf16(k+1)}
// -------------------------------------------------------------------------
__global__ void k_pack(const float* __restrict__ Wg, const float* __restrict__ Wv,
                       const float* __restrict__ Wq, const float* __restrict__ Wro,
                       unsigned int* __restrict__ wcP, unsigned int* __restrict__ wroP) {
  int idx = blockIdx.x * 256 + threadIdx.x;
  if (idx < 24576) {
    int i = idx & 7; int t = idx >> 3; int lane = t & 31; t >>= 5;
    int nb = t % 3, kb = t / 3;
    int n = nb * 16 + (lane & 15), kg = lane >> 4;
    int k = kb * 32 + kbase(i, kg);
    const float* src = (n < 8) ? (Wg + (size_t)n * D_)
                     : (n < 40) ? (Wv + (size_t)(n - 8) * D_)
                                : (Wq + (size_t)(n - 40) * D_);
    wcP[idx] = pack2(f2bf(src[k]), f2bf(src[k + 1]));
  }
  if (idx < 131072) {
    int i = idx & 7; int t = idx >> 3; int lane = t & 31; t >>= 5;
    int nb = t & 63, kb = t >> 6;
    int n = nb * 16 + (lane & 15), kg = lane >> 4;
    int k = kb * 32 + kbase(i, kg);
    wroP[idx] = pack2(f2bf(Wro[(size_t)n * F_ + k]), f2bf(Wro[(size_t)n * F_ + k + 1]));
  }
}

// -------------------------------------------------------------------------
// Projection GEMM: (BT x 1024) x (1024 x 48) via v_wmma_f32_16x16x32_bf16.
// One wave per 16-token tile; 3 accumulator tiles (N=48).
// Output: pbuf(BT,48) token-major: [0,8)=gate logits, [8,40)=write_val,
// [40,48)=query logits.  Branch-free stores.
// -------------------------------------------------------------------------
__global__ void __launch_bounds__(256) k_proj(const float* __restrict__ x,
                                              const unsigned int* __restrict__ wcP,
                                              float* __restrict__ pbuf) {
  int lane = threadIdx.x & 31;
  int wave = threadIdx.x >> 5;
  int tile = blockIdx.x * 8 + wave;     // 0..4095
  int t0 = tile * 16;
  int m = lane & 15, kg = lane >> 4;
  const float* xrow = x + (size_t)(t0 + m) * D_;
  v8f acc0 = {}, acc1 = {}, acc2 = {};
  for (int kb = 0; kb < D_ / 32; ++kb) {
    V16 a;
#pragma unroll
    for (int i = 0; i < 8; ++i) {
      int k = kb * 32 + kbase(i, kg);
      float2 f = *(const float2*)(xrow + k);
      a.s[2 * i]     = f2bf(f.x);
      a.s[2 * i + 1] = f2bf(f.y);
    }
    const unsigned int* bp = wcP + ((size_t)(kb * 3) * 32 + lane) * 8;
    V16 b0, b1, b2;
#pragma unroll
    for (int w = 0; w < 8; ++w) {
      b0.u[w] = bp[w]; b1.u[w] = bp[256 + w]; b2.u[w] = bp[512 + w];
    }
    acc0 = __builtin_amdgcn_wmma_f32_16x16x32_bf16(false, a.v, false, b0.v, (short)0, acc0, false, false);
    acc1 = __builtin_amdgcn_wmma_f32_16x16x32_bf16(false, a.v, false, b1.v, (short)0, acc1, false, false);
    acc2 = __builtin_amdgcn_wmma_f32_16x16x32_bf16(false, a.v, false, b2.v, (short)0, acc2, false, false);
  }
  V8F r0, r1, r2; r0.v = acc0; r1.v = acc1; r2.v = acc2;
  int ncol = lane & 15;
#pragma unroll
  for (int r = 0; r < 8; ++r) {
    size_t row = (size_t)t0 + r + kg * 8;
    pbuf[row * P_ + ncol]      = r0.f[r];
    pbuf[row * P_ + 16 + ncol] = r1.f[r];
    pbuf[row * P_ + 32 + ncol] = r2.f[r];
  }
}

// -------------------------------------------------------------------------
// Scan phase A: per (batch, chunk) compute composed affine map (A, Bv)
// for each of the 256 state elements (s,c).  a_t = 1-sigmoid(g), b_t = w*v.
// -------------------------------------------------------------------------
__global__ void __launch_bounds__(256) k_scan_chunks(const float* __restrict__ pbuf,
                                                     float* __restrict__ chA,
                                                     float* __restrict__ chB) {
  int b = blockIdx.x, ch = blockIdx.y;
  int s = threadIdx.x >> 5, c = threadIdx.x & 31;
  size_t tok = (size_t)b * T_ + (size_t)ch * CH;
  float A = 1.f, Bv = 0.f;
  for (int i = 0; i < CH; ++i, ++tok) {
    float w  = 1.f / (1.f + __expf(-pbuf[tok * P_ + s]));
    float vv = pbuf[tok * P_ + 8 + c];
    float a  = 1.f - w;
    Bv = a * Bv + w * vv;
    A *= a;
  }
  size_t cidx = ((size_t)b * NCH + ch) * S_ + s;
  if (c == 0) chA[cidx] = A;
  chB[cidx * C_ + c] = Bv;
}

// Scan phase B: serial carry across the 32 chunks; emits chunk-start states
// and the `final` memory output.
__global__ void __launch_bounds__(256) k_scan_carry(const float* __restrict__ mem0,
                                                    const float* __restrict__ chA,
                                                    const float* __restrict__ chB,
                                                    float* __restrict__ chS,
                                                    float* __restrict__ outF) {
  int b = blockIdx.x;
  int s = threadIdx.x >> 5, c = threadIdx.x & 31;
  float cur = mem0[((size_t)b * S_ + s) * C_ + c];
  for (int j = 0; j < NCH; ++j) {
    size_t cidx = ((size_t)b * NCH + j) * S_ + s;
    chS[cidx * C_ + c] = cur;
    cur = chA[cidx] * cur + chB[cidx * C_ + c];
  }
  outF[((size_t)b * S_ + s) * C_ + c] = cur;
}

// -------------------------------------------------------------------------
// Scan phase C: replay each chunk from its start state, apply softmax read
// weights, and emit read_flat directly in WMMA A-FRAGMENT layout (bf16):
//   rfA[ ((tile*8 + kb)*32 + lane)*16 + (2*i + half) ]
// where for state element f = s*32+c: kb=s, half=c&1, e=c>>1,
// i=(e&3)+4*(e>=8), kg=(e>>2)&1, lane=(tok&15)+16*kg.
// -------------------------------------------------------------------------
__global__ void __launch_bounds__(256) k_scan_emit(const float* __restrict__ pbuf,
                                                   const float* __restrict__ chS,
                                                   unsigned short* __restrict__ rfA) {
  int b = blockIdx.x, ch = blockIdx.y;
  int s = threadIdx.x >> 5, c = threadIdx.x & 31;
  size_t cidx = ((size_t)b * NCH + ch) * S_ + s;
  float cur = chS[cidx * C_ + c];
  // invert A-fragment layout for this (s,c)
  int half = c & 1;
  int e = c >> 1;
  int i = (e & 3) + ((e >= 8) ? 4 : 0);
  int kg = (e >> 2) & 1;
  int elem = 2 * i + half;             // 0..15 within the lane's fragment
  size_t tok = (size_t)b * T_ + (size_t)ch * CH;
  for (int it = 0; it < CH; ++it, ++tok) {
    float w  = 1.f / (1.f + __expf(-pbuf[tok * P_ + s]));
    float vv = pbuf[tok * P_ + 8 + c];
    cur = (1.f - w) * cur + w * vv;
    float mx = -1e30f;
#pragma unroll
    for (int j = 0; j < 8; ++j) mx = fmaxf(mx, pbuf[tok * P_ + 40 + j]);
    float sum = 0.f;
#pragma unroll
    for (int j = 0; j < 8; ++j) sum += __expf(pbuf[tok * P_ + 40 + j] - mx);
    float rw = __expf(pbuf[tok * P_ + 40 + s] - mx) / sum;
    size_t tile = tok >> 4;
    int lane = (int)(tok & 15) + 16 * kg;
    rfA[(((tile * 8 + s) * 32 + lane) << 4) + elem] = f2bf(rw * cur);
  }
}

// -------------------------------------------------------------------------
// Output GEMM: (BT x 256) x (256 x 1024) via v_wmma_f32_16x16x32_bf16.
// One block per 16-token tile; 8 waves each cover N=128 (8 C tiles).
// A panel (8KB, already fragment-ordered) is staged through LDS once and
// reread by all 8 waves with ds_load_b128.
// -------------------------------------------------------------------------
__global__ void __launch_bounds__(256) k_out(const unsigned int* __restrict__ rfA,
                                             const unsigned int* __restrict__ wroP,
                                             float* __restrict__ out) {
  __shared__ __align__(16) unsigned int ldsA[8 * 32 * 8];   // 8 KB
  int tid = threadIdx.x;
  int lane = tid & 31, wave = tid >> 5;
  int tile = blockIdx.x;
  int t0 = tile * 16;
  { // cooperative stage: 256 threads x 32B
    const uint4* src = (const uint4*)(rfA + (size_t)tile * (8 * 32 * 8)) + tid * 2;
    uint4 q0 = src[0], q1 = src[1];
    uint4* dst = (uint4*)ldsA + tid * 2;
    dst[0] = q0; dst[1] = q1;
  }
  __syncthreads();
  int nbase = wave * 128;
  v8f z = {};
  v8f acc[8];
#pragma unroll
  for (int j = 0; j < 8; ++j) acc[j] = z;
  for (int kb = 0; kb < 8; ++kb) {
    V16 a;
    const uint4* ap = (const uint4*)(ldsA + (kb * 32 + lane) * 8);
    uint4 a0 = ap[0], a1 = ap[1];
    a.u[0] = a0.x; a.u[1] = a0.y; a.u[2] = a0.z; a.u[3] = a0.w;
    a.u[4] = a1.x; a.u[5] = a1.y; a.u[6] = a1.z; a.u[7] = a1.w;
#pragma unroll
    for (int j = 0; j < 8; ++j) {
      const uint4* b4 = (const uint4*)(wroP + (((size_t)kb * 64 + (nbase >> 4) + j) * 32 + lane) * 8);
      uint4 q0 = b4[0], q1 = b4[1];
      V16 bm;
      bm.u[0] = q0.x; bm.u[1] = q0.y; bm.u[2] = q0.z; bm.u[3] = q0.w;
      bm.u[4] = q1.x; bm.u[5] = q1.y; bm.u[6] = q1.z; bm.u[7] = q1.w;
      acc[j] = __builtin_amdgcn_wmma_f32_16x16x32_bf16(false, a.v, false, bm.v, (short)0, acc[j], false, false);
    }
  }
  int kg = lane >> 4;
#pragma unroll
  for (int j = 0; j < 8; ++j) {
    V8F r; r.v = acc[j];
    int n = nbase + j * 16 + (lane & 15);
#pragma unroll
    for (int rr = 0; rr < 8; ++rr) {
      size_t row = (size_t)t0 + rr + kg * 8;
      out[row * D_ + n] = r.f[rr];
    }
  }
}

// -------------------------------------------------------------------------
extern "C" void kernel_launch(void* const* d_in, const int* in_sizes, int n_in,
                              void* d_out, int out_size, void* d_ws, size_t ws_size,
                              hipStream_t stream) {
  const float* x    = (const float*)d_in[0];
  const float* mem0 = (const float*)d_in[1];
  const float* Wg   = (const float*)d_in[2];
  const float* Wv   = (const float*)d_in[3];
  const float* Wq   = (const float*)d_in[4];
  const float* Wro  = (const float*)d_in[5];

  float* out       = (float*)d_out;
  float* out_final = out + (size_t)B_ * T_ * D_;

  char* ws = (char*)d_ws;
  auto carve = [&](size_t bytes) -> void* {
    void* p = (void*)ws;
    ws += (bytes + 255) & ~(size_t)255;
    return p;
  };
  float*          pbuf = (float*)carve((size_t)BT * P_ * 4);          // 12 MB
  unsigned short* rfA  = (unsigned short*)carve((size_t)BT * F_ * 2); // 32 MB
  unsigned int*   wcP  = (unsigned int*)carve((size_t)24576 * 4);     // 96 KB
  unsigned int*   wroP = (unsigned int*)carve((size_t)131072 * 4);    // 512 KB
  float*          chA  = (float*)carve((size_t)B_ * NCH * S_ * 4);
  float*          chB  = (float*)carve((size_t)B_ * NCH * S_ * C_ * 4);
  float*          chS  = (float*)carve((size_t)B_ * NCH * S_ * C_ * 4);

  k_pack<<<512, 256, 0, stream>>>(Wg, Wv, Wq, Wro, wcP, wroP);
  k_proj<<<(int)(BT / 16 / 8), 256, 0, stream>>>(x, wcP, pbuf);
  k_scan_chunks<<<dim3(B_, NCH), 256, 0, stream>>>(pbuf, chA, chB);
  k_scan_carry<<<B_, 256, 0, stream>>>(mem0, chA, chB, chS, out_final);
  k_scan_emit<<<dim3(B_, NCH), 256, 0, stream>>>(pbuf, chS, rfA);
  k_out<<<(int)(BT / 16), 256, 0, stream>>>((const unsigned int*)rfA, wroP, out);
}
